// SimplifiedMultiHeadAttention_34505767256262
// MI455X (gfx1250) — matrix operand order repaired and checked by reading, not compiled, not run on hardware
//
#include <hip/hip_runtime.h>
#include <hip/hip_bf16.h>
#include <math.h>

// ---------------------------------------------------------------------------
// SimplifiedMultiHeadAttention for MI455X (gfx1250, wave32, WMMA).
//
// Roofline: ~34 GFLOP vs tens of MB of traffic -> strongly compute bound.
// v_wmma_f32_16x16x32_bf16 (bf16 in, fp32 accumulate) for all GEMMs.
// Tile staging per K-step uses all three CDNA5 data paths:
//   A tiles     : TDM tensor_load_to_lds (TENSORcnt), one 2D tile per wave,
//                 with D# pad fields producing the bank-skewed LDS layout.
//   B tiles (NN): global_load_async_to_lds_b128 (ASYNCcnt).
//   B tiles (TN): VALU transpose staging (DMA cannot transpose).
// Double-buffered LDS overlaps the next tile's fill with current WMMAs.
// ---------------------------------------------------------------------------

#define SEQ 2048
#define DM  1024
#define NH  16
#define DK  64

#define AS_LD 40   // A tile row stride (elements): 32 + 8 pad (TDM pad fields)
#define BS_LD 72   // B tile row stride (elements): 64 + 8 pad

typedef __bf16 v16bf __attribute__((ext_vector_type(16)));
typedef __bf16 v8bf  __attribute__((ext_vector_type(8)));
typedef float  v8f   __attribute__((ext_vector_type(8)));
typedef int    v4i   __attribute__((ext_vector_type(4)));
typedef int    v8i   __attribute__((ext_vector_type(8)));

__device__ __forceinline__ unsigned short f32_to_bf16(float f) {
  union { float f; unsigned u; } x; x.f = f;
  unsigned r = x.u + 0x7FFFu + ((x.u >> 16) & 1u);   // round-to-nearest-even
  return (unsigned short)(r >> 16);
}
__device__ __forceinline__ float bf16_to_f32(unsigned short h) {
  union { unsigned u; float f; } x; x.u = ((unsigned)h) << 16; return x.f;
}

__device__ __forceinline__ unsigned lds_off_of(const void* p) {
  // Generic pointers to LDS carry the workgroup-relative offset in the low
  // 32 bits (shared aperture occupies the high bits).
  return (unsigned)(unsigned long long)p;
}

// ---- gfx1250 async copy: 16 B global -> LDS, tracked by ASYNCcnt ----------
__device__ __forceinline__ void async_copy_b128(unsigned lds_off, const void* gaddr) {
  asm volatile("global_load_async_to_lds_b128 %0, %1, off"
               :: "v"(lds_off), "v"(gaddr) : "memory");
}
__device__ __forceinline__ void wait_async0() {
  asm volatile("s_wait_asynccnt 0" ::: "memory");
}
__device__ __forceinline__ void wait_tensor0() {
  __builtin_amdgcn_s_wait_tensorcnt(0);
}

// ---- gfx1250 Tensor Data Mover: 2D tile load, bf16 elements ---------------
// D# per CDNA5 ISA 8.3/8.4: group0 = {flags/count, lds_addr, global_addr,
// type=2}; group1 = {data_size/pads, tensor dims, tile dims, dim0 stride}.
// pad_interval/pad_amount skip LDS bytes after each row (bank skew); padding
// advances the LDS pointer only (no writes), so no overrun past the tile.
__device__ __forceinline__ void tdm_load_2d_bf16(
    unsigned lds_byte_off, const void* gaddr,
    unsigned tile_d0, unsigned tile_d1,
    unsigned long long stride_elems,
    unsigned pad_interval_code, unsigned pad_amount_code) {
  const unsigned long long ga = (unsigned long long)gaddr;
  v4i g0;
  g0[0] = 1;                                              // count=1, user D#
  g0[1] = (int)lds_byte_off;                              // lds_addr
  g0[2] = (int)(unsigned)(ga & 0xFFFFFFFFu);              // global_addr[31:0]
  g0[3] = (int)((unsigned)((ga >> 32) & 0x01FFFFFFu) | (2u << 30)); // +type=2
  v8i g1;
  g1[0] = (int)((1u << 16) |                              // data_size=1 (2 B)
                (1u << 20) |                              // pad_enable
                (pad_interval_code << 22) |
                (pad_amount_code << 25));
  g1[1] = (int)((tile_d0 & 0xFFFFu) << 16);               // tensor_dim0 lo16
  g1[2] = (int)((tile_d0 >> 16) | ((tile_d1 & 0xFFFFu) << 16)); // d0 hi|d1 lo
  g1[3] = (int)((tile_d1 >> 16) | (tile_d0 << 16));       // d1 hi | tile_dim0
  g1[4] = (int)(tile_d1 & 0xFFFFu);                       // tile_dim1; dim2=0
  g1[5] = (int)(unsigned)(stride_elems & 0xFFFFFFFFu);    // dim0 stride lo32
  g1[6] = (int)(unsigned)((stride_elems >> 32) & 0xFFFFu);// dim0 stride hi16
  g1[7] = 0;                                              // dim1 stride (2D)
  asm volatile("tensor_load_to_lds %0, %1" :: "s"(g0), "s"(g1) : "memory");
}

// ---------------------------------------------------------------------------
// fp32 -> bf16 elementwise conversion
// ---------------------------------------------------------------------------
__global__ void cvt_f32_bf16(const float* __restrict__ src,
                             unsigned short* __restrict__ dst, int n) {
  int i = blockIdx.x * blockDim.x + threadIdx.x;
  if (i < n) dst[i] = f32_to_bf16(src[i]);
}

// ---------------------------------------------------------------------------
// Tiled WMMA GEMM, double-buffered TDM/async staging.
//   TRANSB = true : C[m,n] = sum_k A[m,k] * B[n,k]   (TN; both K-contiguous)
//   TRANSB = false: C[m,n] = sum_k A[m,k] * B[k,n]   (NN)
// Block = 256 threads = 8 waves; block tile 128x64; each wave owns a 16x64
// strip (4 f32 accumulators). K stepped by 32. Output: (acc + bias[col]) *
// scale, stored bf16 (Cb) or f32 (Cf). M % 128 == 0, N % 64 == 0,
// Kdim % 32 == 0 for all calls here.
// ---------------------------------------------------------------------------
template <bool TRANSB>
__global__ __launch_bounds__(256)
void gemm_bf16_wmma(const unsigned short* __restrict__ A, int lda,
                    const unsigned short* __restrict__ B, int ldb,
                    unsigned short* __restrict__ Cb,
                    float* __restrict__ Cf, int ldc,
                    int Kdim, const float* __restrict__ bias, float scale) {
  __shared__ alignas(16) unsigned short As[2][128 * AS_LD];  // [row][k]
  __shared__ alignas(16) unsigned short Bs[2][32 * BS_LD];   // [k][n]

  const int tid  = threadIdx.x;
  const int wave = tid >> 5;
  const int lane = tid & 31;
  const int l16  = lane & 15;
  const int hi   = lane >> 4;           // 0: lanes 0-15, 1: lanes 16-31

  const int mBase = blockIdx.y * 128;
  const int nBase = blockIdx.x * 64;

  // --- A tile: one TDM 2D tile per wave (its own 16-row strip).
  // Rows of 32 bf16 = 16 DWORDs -> pad_interval code 3; skip 4 DWORDs
  // (8 elements) after each row -> pad_amount code 3; LDS stride = AS_LD.
  auto stageA = [&](int kb, int buf) {
    tdm_load_2d_bf16(lds_off_of(&As[buf][(wave * 16) * AS_LD]),
                     A + (size_t)(mBase + wave * 16) * lda + kb,
                     /*tile_d0=*/32, /*tile_d1=*/16,
                     (unsigned long long)lda,
                     /*interval=*/3, /*amount=*/3);
  };
  auto stageB = [&](int kb, int buf) {
    if (TRANSB) {
      // B[n][k] -> Bs[k][n]: transpose via VGPRs (DMA cannot transpose)
      const int n = tid >> 2, k8 = (tid & 3) << 3;       // 256 chunks of 8
      union { v8bf v; unsigned short s[8]; } t;
      t.v = *reinterpret_cast<const v8bf*>(B + (size_t)(nBase + n) * ldb + kb + k8);
      #pragma unroll
      for (int j = 0; j < 8; ++j) Bs[buf][(k8 + j) * BS_LD + n] = t.s[j];
    } else {
      // B[k][n] -> Bs[k][n]: async global->LDS b128 copies
      const int k = tid >> 3, n8 = (tid & 7) << 3;       // 256 chunks of 8
      async_copy_b128(lds_off_of(&Bs[buf][k * BS_LD + n8]),
                      B + (size_t)(kb + k) * ldb + nBase + n8);
    }
  };

  const v8f zero = {0.f, 0.f, 0.f, 0.f, 0.f, 0.f, 0.f, 0.f};
  v8f acc[4] = {zero, zero, zero, zero};

  // prologue: fill buffer 0
  stageA(0, 0);
  stageB(0, 0);
  wait_tensor0();
  wait_async0();
  __syncthreads();

  int buf = 0;
  for (int kb = 0; kb < Kdim; kb += 32) {
    // kick off next tile into the other buffer while we compute
    if (kb + 32 < Kdim) {
      stageA(kb + 32, buf ^ 1);
      stageB(kb + 32, buf ^ 1);
      __builtin_prefetch(A + (size_t)(mBase + tid / 2) * lda + kb + 64 + (tid & 1) * 16, 0, 1);
    }

    // A fragment (ISA 16-bit A layout: lane<16 -> K 0-7,16-23; lane>=16 -> K 8-15,24-31)
    union { v16bf v; v8bf h[2]; } afr;
    const int r = wave * 16 + l16;
    afr.h[0] = *reinterpret_cast<const v8bf*>(&As[buf][r * AS_LD + hi * 8]);
    afr.h[1] = *reinterpret_cast<const v8bf*>(&As[buf][r * AS_LD + hi * 8 + 16]);

    // 4 WMMA ops (N subtiles); B fragment: lane&15 = n, K in same order as A
    #pragma unroll
    for (int j = 0; j < 4; ++j) {
      v16bf bfr;
      #pragma unroll
      for (int s = 0; s < 16; ++s) {
        const int kk = (s < 8) ? (hi * 8 + s) : (16 + hi * 8 + (s - 8));
        bfr[s] = __builtin_bit_cast(__bf16, Bs[buf][kk * BS_LD + j * 16 + l16]);
      }
      acc[j] = __builtin_amdgcn_wmma_f32_16x16x32_bf16(
          false, afr.v, false, bfr, (short)0, acc[j], false, false);
    }

    wait_tensor0();    // my next-buffer TDM tile landed ...
    wait_async0();     // ... and my async b128 shares ...
    __syncthreads();   // ... and everyone done reading current buffer
    buf ^= 1;
  }

  // writeout: C layout: element rr, lane L -> M = rr + 8*(L>=16), N = L&15
  const int mOut = mBase + wave * 16 + hi * 8;
  #pragma unroll
  for (int j = 0; j < 4; ++j) {
    const int col = nBase + j * 16 + l16;
    const float b = bias ? bias[col] : 0.f;
    #pragma unroll
    for (int rr = 0; rr < 8; ++rr) {
      float v = (acc[j][rr] + b) * scale;
      if (Cf) Cf[(size_t)(mOut + rr) * ldc + col] = v;
      else    Cb[(size_t)(mOut + rr) * ldc + col] = f32_to_bf16(v);
    }
  }
}

// ---------------------------------------------------------------------------
// Row softmax over bf16 scores, in place. One block per row, fp32 math.
// ---------------------------------------------------------------------------
__global__ __launch_bounds__(256)
void softmax_rows_bf16(unsigned short* __restrict__ S, int n) {
  __shared__ float red[256];
  const int tid = threadIdx.x;
  unsigned short* p = S + (size_t)blockIdx.x * n;

  float lmax = -3.402823e38f;
  for (int i = tid; i < n; i += 256) lmax = fmaxf(lmax, bf16_to_f32(p[i]));
  red[tid] = lmax; __syncthreads();
  for (int s = 128; s > 0; s >>= 1) {
    if (tid < s) red[tid] = fmaxf(red[tid], red[tid + s]);
    __syncthreads();
  }
  const float m = red[0];
  __syncthreads();

  float lsum = 0.f;
  for (int i = tid; i < n; i += 256) {
    float e = __expf(bf16_to_f32(p[i]) - m);
    lsum += e;
    p[i] = f32_to_bf16(e);
  }
  red[tid] = lsum; __syncthreads();
  for (int s = 128; s > 0; s >>= 1) {
    if (tid < s) red[tid] += red[tid + s];
    __syncthreads();
  }
  const float inv = 1.f / red[0];
  __syncthreads();
  for (int i = tid; i < n; i += 256) p[i] = f32_to_bf16(bf16_to_f32(p[i]) * inv);
}

// ---------------------------------------------------------------------------
// Host orchestration (graph-capture safe: launches only, all on `stream`).
// Workspace layout (bf16 elements), total 36 MB:
//   xb[S*D] Wqb,Wkb,Wvb,Wob[D*D] Qb,Kb,Vb[S*D] Attb[S*D] Sc[S*S]
// ---------------------------------------------------------------------------
extern "C" void kernel_launch(void* const* d_in, const int* in_sizes, int n_in,
                              void* d_out, int out_size, void* d_ws, size_t ws_size,
                              hipStream_t stream) {
  (void)in_sizes; (void)n_in; (void)out_size; (void)ws_size;

  const float* x  = (const float*)d_in[0];
  const float* Wq = (const float*)d_in[1]; const float* bq = (const float*)d_in[2];
  const float* Wk = (const float*)d_in[3]; const float* bk = (const float*)d_in[4];
  const float* Wv = (const float*)d_in[5]; const float* bv = (const float*)d_in[6];
  const float* Wo = (const float*)d_in[7]; const float* bo = (const float*)d_in[8];
  float* out = (float*)d_out;

  unsigned short* xb   = (unsigned short*)d_ws;
  unsigned short* Wqb  = xb   + (size_t)SEQ * DM;
  unsigned short* Wkb  = Wqb  + (size_t)DM * DM;
  unsigned short* Wvb  = Wkb  + (size_t)DM * DM;
  unsigned short* Wob  = Wvb  + (size_t)DM * DM;
  unsigned short* Qb   = Wob  + (size_t)DM * DM;
  unsigned short* Kb   = Qb   + (size_t)SEQ * DM;
  unsigned short* Vb   = Kb   + (size_t)SEQ * DM;
  unsigned short* Attb = Vb   + (size_t)SEQ * DM;
  unsigned short* Sc   = Attb + (size_t)SEQ * DM;   // one head's 2048x2048 scores

  // fp32 -> bf16 conversions
  cvt_f32_bf16<<<(SEQ * DM) / 256, 256, 0, stream>>>(x,  xb,  SEQ * DM);
  cvt_f32_bf16<<<(DM * DM) / 256, 256, 0, stream>>>(Wq, Wqb, DM * DM);
  cvt_f32_bf16<<<(DM * DM) / 256, 256, 0, stream>>>(Wk, Wkb, DM * DM);
  cvt_f32_bf16<<<(DM * DM) / 256, 256, 0, stream>>>(Wv, Wvb, DM * DM);
  cvt_f32_bf16<<<(DM * DM) / 256, 256, 0, stream>>>(Wo, Wob, DM * DM);

  const dim3 blk(256);
  const dim3 gProj(DM / 64, SEQ / 128);

  // Q/K/V projections; 1/sqrt(dk)=0.125 folded into Q.
  gemm_bf16_wmma<true><<<gProj, blk, 0, stream>>>(xb, DM, Wqb, DM, Qb, nullptr, DM, DM, bq, 0.125f);
  gemm_bf16_wmma<true><<<gProj, blk, 0, stream>>>(xb, DM, Wkb, DM, Kb, nullptr, DM, DM, bk, 1.0f);
  gemm_bf16_wmma<true><<<gProj, blk, 0, stream>>>(xb, DM, Wvb, DM, Vb, nullptr, DM, DM, bv, 1.0f);

  const dim3 gScore(SEQ / 64, SEQ / 128);
  const dim3 gPV(DK / 64, SEQ / 128);   // (1, 16)

  for (int h = 0; h < NH; ++h) {
    // scores_h = Q_h @ K_h^T  (TN, K-dim = dk = 64)
    gemm_bf16_wmma<true><<<gScore, blk, 0, stream>>>(
        Qb + h * DK, DM, Kb + h * DK, DM, Sc, nullptr, SEQ, DK, nullptr, 1.0f);
    // softmax rows
    softmax_rows_bf16<<<SEQ, 256, 0, stream>>>(Sc, SEQ);
    // attn_h = P @ V_h  (NN, K-dim = S = 2048), written into Attb columns
    gemm_bf16_wmma<false><<<gPV, blk, 0, stream>>>(
        Sc, SEQ, Vb + h * DK, DM, Attb + h * DK, nullptr, DM, SEQ, nullptr, 1.0f);
  }

  // out = Attb @ Wo^T + bo  (fp32 result)
  gemm_bf16_wmma<true><<<gProj, blk, 0, stream>>>(
      Attb, DM, Wob, DM, nullptr, out, DM, DM, bo, 1.0f);
}
